// GatedLSTM_34179349742312
// MI455X (gfx1250) — matrix operand order repaired and checked by reading, not compiled, not run on hardware
//
#include <hip/hip_runtime.h>
#include <hip/hip_bf16.h>
#include <cstdint>

// GatedLSTM: B=64, S=512, I=1024, H=1024
// Gate "i" is computed-but-unused in the reference -> skipped entirely.
#define B_ 64
#define S_ 512
#define I_ 1024
#define H_ 1024

typedef __attribute__((ext_vector_type(16))) __bf16 v16bf;
typedef __attribute__((ext_vector_type(8)))  __bf16 v8bf;
typedef __attribute__((ext_vector_type(8)))  float  v8f;

#define WMMA_BF16(A, Bm, C) \
  __builtin_amdgcn_wmma_f32_16x16x32_bf16(false, (A), false, (Bm), (short)0, (C), false, false)

// ---------------------------------------------------------------------------
// Fragment loaders (wave32, WMMA 16x16x32 bf16; contiguous K order, same for A and B)
// A: 16x32 (MxK). lane L: M = L&15; halves 0..15 = K (k0+base..+15),
//    base = (L<16 ? 0 : 16).
// ---------------------------------------------------------------------------
__device__ __forceinline__ v16bf load_a_frag_bf16(const __bf16* __restrict__ base,
                                                  long lda, int lane, int k0) {
  const int m    = lane & 15;
  const int koff = k0 + ((lane >> 4) << 4);   // +0 or +16
  const __bf16* row = base + (long)m * lda + koff;
  const v8bf lo = *reinterpret_cast<const v8bf*>(row);
  const v8bf hi = *reinterpret_cast<const v8bf*>(row + 8);
  return __builtin_shufflevector(lo, hi, 0, 1, 2, 3, 4, 5, 6, 7,
                                 8, 9, 10, 11, 12, 13, 14, 15);
}

// B: 32x16 (KxN), B[k][n] = Wt[n][k] (weight row n is a B column, K-contiguous).
// lane L: N = L&15; K = k0 + (L<16 ? 0..15 : 16..31)  -> one 32B contiguous read.
__device__ __forceinline__ v16bf load_b_frag_bf16(const __bf16* __restrict__ Wt,
                                                  long ldw, int lane, int n0, int k0) {
  const int n  = n0 + (lane & 15);
  const int kb = k0 + ((lane >> 4) << 4);     // +0 or +16
  return *reinterpret_cast<const v16bf*>(Wt + (long)n * ldw + kb);
}

__device__ __forceinline__ float sigmoidf_(float x) {
  return 1.0f / (1.0f + __expf(-x));
}

// Fragment set for one K-chunk: 2 A tiles + 3 gate B tiles.
struct Frags {
  v16bf a0, a1, f, o, c;
};

__device__ __forceinline__ Frags load_frags(const __bf16* __restrict__ A0,
                                            const __bf16* __restrict__ A1, long lda,
                                            const __bf16* __restrict__ Bf,
                                            const __bf16* __restrict__ Bo,
                                            const __bf16* __restrict__ Bc, long ldb,
                                            int lane, int n0, int k0) {
  Frags r;
  r.a0 = load_a_frag_bf16(A0, lda, lane, k0);
  r.a1 = load_a_frag_bf16(A1, lda, lane, k0);
  r.f  = load_b_frag_bf16(Bf, ldb, lane, n0, k0);
  r.o  = load_b_frag_bf16(Bo, ldb, lane, n0, k0);
  r.c  = load_b_frag_bf16(Bc, ldb, lane, n0, k0);
  return r;
}

__device__ __forceinline__ void mma6(const Frags& fr, v8f accf[2], v8f acco[2],
                                     v8f accc[2]) {
  accf[0] = WMMA_BF16(fr.a0, fr.f, accf[0]);
  accf[1] = WMMA_BF16(fr.a1, fr.f, accf[1]);
  acco[0] = WMMA_BF16(fr.a0, fr.o, acco[0]);
  acco[1] = WMMA_BF16(fr.a1, fr.o, acco[1]);
  accc[0] = WMMA_BF16(fr.a0, fr.c, accc[0]);
  accc[1] = WMMA_BF16(fr.a1, fr.c, accc[1]);
}

// ---------------------------------------------------------------------------
// f32 -> bf16 conversion (weights and X), vectorized 4-wide
// ---------------------------------------------------------------------------
__global__ __launch_bounds__(256)
void cvt_f32_bf16_x4(const float* __restrict__ src, __bf16* __restrict__ dst, int n4) {
  int i = blockIdx.x * blockDim.x + threadIdx.x;
  if (i < n4) {
    const float4 v = reinterpret_cast<const float4*>(src)[i];
    __bf16 r0 = (__bf16)v.x, r1 = (__bf16)v.y, r2 = (__bf16)v.z, r3 = (__bf16)v.w;
    ((ushort*)dst)[4 * i + 0] = __builtin_bit_cast(ushort, r0);
    ((ushort*)dst)[4 * i + 1] = __builtin_bit_cast(ushort, r1);
    ((ushort*)dst)[4 * i + 2] = __builtin_bit_cast(ushort, r2);
    ((ushort*)dst)[4 * i + 3] = __builtin_bit_cast(ushort, r3);
  }
}

// ---------------------------------------------------------------------------
// Phase 1: fused input projections  xg = X @ W_g^T + b_g   (g in {f,o,c})
// 1 wave = 32x16 output (2 M-tiles share the 3 B fragments).
// grid: 1024 M-groups x 16 N-groups, 128 threads (4 waves).
// Ping/pong double-buffered, unrolled x2 -> zero register-copy swaps.
// ---------------------------------------------------------------------------
__global__ __launch_bounds__(128)
void lstm_xproj_kernel(const __bf16* __restrict__ Xb,
                       const __bf16* __restrict__ Wf, const __bf16* __restrict__ Wo,
                       const __bf16* __restrict__ Wc,
                       const float* __restrict__ bfp, const float* __restrict__ bop,
                       const float* __restrict__ bcp,
                       float* __restrict__ xf, float* __restrict__ xo,
                       float* __restrict__ xc) {
  const int lane = threadIdx.x & 31;
  const int wave = threadIdx.x >> 5;
  const int mg   = blockIdx.x >> 4;                  // 0..1023  (32 rows each)
  const int nt   = ((blockIdx.x & 15) << 2) + wave;  // 0..63    (H/16)
  const int n0   = nt * 16;

  const __bf16* A0 = Xb + (long)mg * 32 * I_;
  const __bf16* A1 = A0 + (long)16 * I_;

  v8f accf[2] = {{}, {}}, acco[2] = {{}, {}}, accc[2] = {{}, {}};

  Frags p = load_frags(A0, A1, I_, Wf, Wo, Wc, I_, lane, n0, 0);
  Frags q;

#pragma unroll 1
  for (int k0 = 0; k0 < I_ - 64; k0 += 64) {
    q = load_frags(A0, A1, I_, Wf, Wo, Wc, I_, lane, n0, k0 + 32);
    mma6(p, accf, acco, accc);
    p = load_frags(A0, A1, I_, Wf, Wo, Wc, I_, lane, n0, k0 + 64);
    mma6(q, accf, acco, accc);
  }
  q = load_frags(A0, A1, I_, Wf, Wo, Wc, I_, lane, n0, I_ - 32);
  mma6(p, accf, acco, accc);
  mma6(q, accf, acco, accc);

  // C/D layout: VGPR j -> M = j + (lane<16?0:8), N = lane&15
  const int n     = n0 + (lane & 15);
  const float bfv = bfp[n], bov = bop[n], bcv = bcp[n];
#pragma unroll
  for (int half = 0; half < 2; ++half) {
    const int mbase = mg * 32 + half * 16 + ((lane >> 4) << 3);
#pragma unroll
    for (int j = 0; j < 8; ++j) {
      const long idx = (long)(mbase + j) * H_ + n;
      xf[idx] = accf[half][j] + bfv;
      xo[idx] = acco[half][j] + bov;
      xc[idx] = accc[half][j] + bcv;
    }
  }
}

// ---------------------------------------------------------------------------
// Phase 2: one recurrent step.  h_{t-1} read from bf16 ping buffer,
// gates = sigmoid/tanh(x_pre + h @ U^T); writes h_t (f32) into out[:, t, :]
// and h_t (bf16) into the pong buffer for the next step's A fragments.
// 1 wave = 32x16 output; grid: 32 blocks x 128 threads = 128 waves.
// ---------------------------------------------------------------------------
__global__ __launch_bounds__(128)
void lstm_step_kernel(const __bf16* __restrict__ Uf, const __bf16* __restrict__ Uo,
                      const __bf16* __restrict__ Uc,
                      const float* __restrict__ xf, const float* __restrict__ xo,
                      const float* __restrict__ xc,
                      const __bf16* __restrict__ hprev, __bf16* __restrict__ hnext,
                      float* __restrict__ out, float* __restrict__ cbuf, int t) {
  const int lane = threadIdx.x & 31;
  const int wave = threadIdx.x >> 5;
  const int mg   = blockIdx.x & 1;                      // 0..1  (32 rows each)
  const int nt   = ((blockIdx.x >> 1) << 2) + wave;     // 0..63 (H/16)
  const int n0   = nt * 16;
  const int n    = n0 + (lane & 15);

  // Prefetch the epilogue pre-activation rows (gfx1250 global_prefetch) so the
  // elementwise tail (and the whole t==0 step) doesn't stall on HBM/L2.
#pragma unroll
  for (int half = 0; half < 2; ++half) {
    const int mbase = mg * 32 + half * 16 + ((lane >> 4) << 3);
#pragma unroll
    for (int j = 0; j < 8; ++j) {
      const long xidx = ((long)(mbase + j) * S_ + t) * (long)H_ + n;
      __builtin_prefetch(&xf[xidx], 0, 1);
      __builtin_prefetch(&xo[xidx], 0, 1);
      __builtin_prefetch(&xc[xidx], 0, 1);
    }
  }

  v8f accf[2] = {{}, {}}, acco[2] = {{}, {}}, accc[2] = {{}, {}};

  if (t > 0) {
    const __bf16* A0 = hprev + (long)(mg * 32) * H_;
    const __bf16* A1 = A0 + (long)16 * H_;

    Frags p = load_frags(A0, A1, H_, Uf, Uo, Uc, H_, lane, n0, 0);
    Frags q;

#pragma unroll 1
    for (int k0 = 0; k0 < H_ - 64; k0 += 64) {
      q = load_frags(A0, A1, H_, Uf, Uo, Uc, H_, lane, n0, k0 + 32);
      mma6(p, accf, acco, accc);
      p = load_frags(A0, A1, H_, Uf, Uo, Uc, H_, lane, n0, k0 + 64);
      mma6(q, accf, acco, accc);
    }
    q = load_frags(A0, A1, H_, Uf, Uo, Uc, H_, lane, n0, H_ - 32);
    mma6(p, accf, acco, accc);
    mma6(q, accf, acco, accc);
  }

#pragma unroll
  for (int half = 0; half < 2; ++half) {
    const int mbase = mg * 32 + half * 16 + ((lane >> 4) << 3);
#pragma unroll
    for (int j = 0; j < 8; ++j) {
      const int  b    = mbase + j;
      const long xidx = ((long)b * S_ + t) * (long)H_ + n;
      const long cidx = (long)b * H_ + n;
      const float f  = sigmoidf_(xf[xidx] + accf[half][j]);
      const float o  = sigmoidf_(xo[xidx] + acco[half][j]);
      const float cp = (t > 0) ? cbuf[cidx] : 0.0f;
      const float cn = f * cp + tanhf(xc[xidx] + accc[half][j]);
      const float hn = o * cn;
      cbuf[cidx]  = cn;
      out[xidx]   = hn;                   // output[b, t, :]
      hnext[cidx] = (__bf16)hn;           // bf16 h for next step's A fragments
      if (t == S_ - 1) {
        const long base = (long)B_ * S_ * H_;
        out[base + cidx]                 = hn;  // h [1,B,H]
        out[base + (long)B_ * H_ + cidx] = cn;  // c [1,B,H]
      }
    }
  }
}

// ---------------------------------------------------------------------------
// Host launcher
// input order: input, W_f, U_f, b_f, W_i, U_i, b_i, W_o, U_o, b_o, W_c, U_c, b_c
// ---------------------------------------------------------------------------
extern "C" void kernel_launch(void* const* d_in, const int* in_sizes, int n_in,
                              void* d_out, int out_size, void* d_ws, size_t ws_size,
                              hipStream_t stream) {
  const float* X   = (const float*)d_in[0];
  const float* W_f = (const float*)d_in[1];
  const float* U_f = (const float*)d_in[2];
  const float* b_f = (const float*)d_in[3];
  const float* W_o = (const float*)d_in[7];
  const float* U_o = (const float*)d_in[8];
  const float* b_o = (const float*)d_in[9];
  const float* W_c = (const float*)d_in[10];
  const float* U_c = (const float*)d_in[11];
  const float* b_c = (const float*)d_in[12];
  float* out = (float*)d_out;

  const long HI   = (long)H_ * I_;          // 1M elems
  const long BSH  = (long)B_ * S_ * H_;     // 33.5M elems
  const long BSI  = (long)B_ * S_ * I_;     // 33.5M elems
  const long BH   = (long)B_ * H_;          // 65536 elems

  char* ws = (char*)d_ws;
  __bf16* Wf = (__bf16*)ws;
  __bf16* Wo = Wf + HI;
  __bf16* Wc = Wo + HI;
  __bf16* Uf = Wc + HI;
  __bf16* Uo = Uf + HI;
  __bf16* Uc = Uo + HI;
  __bf16* Xb = Uc + HI;                     // bf16 copy of X
  __bf16* h0 = Xb + BSI;                    // bf16 h ping
  __bf16* h1 = h0 + BH;                     // bf16 h pong
  float*  xf = (float*)(h1 + BH);
  float*  xo = xf + BSH;
  float*  xc = xo + BSH;
  float*  cb = xc + BSH;                    // [B,H] f32 cell state

  // Conversions (f32 -> bf16): weights + X
  {
    const int n4 = (int)(HI / 4);
    const int g  = (n4 + 255) / 256;
    cvt_f32_bf16_x4<<<g, 256, 0, stream>>>(W_f, Wf, n4);
    cvt_f32_bf16_x4<<<g, 256, 0, stream>>>(W_o, Wo, n4);
    cvt_f32_bf16_x4<<<g, 256, 0, stream>>>(W_c, Wc, n4);
    cvt_f32_bf16_x4<<<g, 256, 0, stream>>>(U_f, Uf, n4);
    cvt_f32_bf16_x4<<<g, 256, 0, stream>>>(U_o, Uo, n4);
    cvt_f32_bf16_x4<<<g, 256, 0, stream>>>(U_c, Uc, n4);
    const int nx4 = (int)(BSI / 4);
    cvt_f32_bf16_x4<<<(nx4 + 255) / 256, 256, 0, stream>>>(X, Xb, nx4);
  }

  // Phase 1: fused input-side projections for gates f, o, c
  lstm_xproj_kernel<<<1024 * 16, 128, 0, stream>>>(Xb, Wf, Wo, Wc,
                                                   b_f, b_o, b_c,
                                                   xf, xo, xc);

  // Phase 2: sequential scan; stream ordering provides the grid-wide sync.
  // bf16 h ping-pong: step t reads hbuf[(t-1)&1], writes hbuf[t&1].
  for (int t = 0; t < S_; ++t) {
    __bf16* hprev = (t & 1) ? h0 : h1;      // h[(t-1)&1]
    __bf16* hnext = (t & 1) ? h1 : h0;      // h[t&1]
    lstm_step_kernel<<<32, 128, 0, stream>>>(Uf, Uo, Uc, xf, xo, xc,
                                             hprev, hnext, out, cb, t);
  }
}
